// SimpleMambaLayer_8624294330965
// MI455X (gfx1250) — matrix operand order repaired and checked by reading, not compiled, not run on hardware
//
#include <hip/hip_runtime.h>
#include <hip/hip_bf16.h>

typedef __attribute__((ext_vector_type(16))) _Float16 v16h;
typedef __attribute__((ext_vector_type(8)))  _Float16 v8h;
typedef __attribute__((ext_vector_type(4)))  _Float16 v4h;
typedef __attribute__((ext_vector_type(8)))  float    v8f;
typedef __attribute__((ext_vector_type(4)))  float    v4f;

#define TILE_M 128
#define TILE_N 128
#define TILE_K 32
#define LDS_PAD 8          // rows of 40 halves = 80B: conflict-free b128 fragment reads

#define BATCH   8
#define L_SEQ   4096
#define DCH     1024
#define SEG     64         // number of segments along L
#define SEGLEN  64         // L_SEQ / SEG

// ---------------------------------------------------------------------------
// Tile stagers: global -> registers -> LDS (f16), overloaded on element type.
// f32 path converts to f16 on the fly; f16 path is a straight b128 copy.
// ---------------------------------------------------------------------------
__device__ __forceinline__ void tile_ld(const float* p, int ld, int row0, int k0,
                                        int tid, v4f st[4]) {
#pragma unroll
    for (int u = 0; u < 4; ++u) {
        const int idx = tid + u * 256;
        const int row = idx >> 3, q = idx & 7;             // 128 rows x 8 quads
        st[u] = *(const v4f*)(p + (size_t)(row0 + row) * ld + k0 + q * 4);
    }
}
__device__ __forceinline__ void tile_st(_Float16 (*lds)[TILE_K + LDS_PAD],
                                        int tid, const v4f st[4]) {
#pragma unroll
    for (int u = 0; u < 4; ++u) {
        const int idx = tid + u * 256;
        const int row = idx >> 3, q = idx & 7;
        *(v4h*)&lds[row][q * 4] = __builtin_convertvector(st[u], v4h);
    }
}
__device__ __forceinline__ void tile_ld(const _Float16* p, int ld, int row0, int k0,
                                        int tid, v8h st[2]) {
#pragma unroll
    for (int u = 0; u < 2; ++u) {
        const int idx = tid + u * 256;
        const int row = idx >> 2, o = idx & 3;             // 128 rows x 4 octs
        st[u] = *(const v8h*)(p + (size_t)(row0 + row) * ld + k0 + o * 8);
    }
}
__device__ __forceinline__ void tile_st(_Float16 (*lds)[TILE_K + LDS_PAD],
                                        int tid, const v8h st[2]) {
#pragma unroll
    for (int u = 0; u < 2; ++u) {
        const int idx = tid + u * 256;
        const int row = idx >> 2, o = idx & 3;
        *(v8h*)&lds[row][o * 8] = st[u];
    }
}

template<typename T> struct Stage;
template<> struct Stage<float>    { using ty = v4f; static const int n = 4; };
template<> struct Stage<_Float16> { using ty = v8h; static const int n = 2; };

// ---------------------------------------------------------------------------
// GEMM: Out[m,n] = act( sum_k X[m,k]*W[n,k] + bias[n] )
// Block tile 128x128, 8 waves (2x4); each wave owns 64x32:
// 4 A-frags x 2 B-frags -> 8 WMMAs per 32-wide K step.
// Double-buffered LDS: one barrier per K step, global loads overlap compute.
// Grid: blockIdx.x = N tile (fast, 8 tiles share one A strip -> L2 reuse),
//       blockIdx.y = M tile.
// ---------------------------------------------------------------------------
template<typename AT, typename OT, bool GATED>
__global__ __launch_bounds__(256)
void gemm_f16_wmma(const AT* __restrict__ X,     // [M,K] row-major
                   const float* __restrict__ W,  // [N,K] row-major
                   const float* __restrict__ bias,
                   OT* __restrict__ Out,         // [M,N]
                   int M, int N, int K)
{
    __shared__ _Float16 As[2][TILE_M][TILE_K + LDS_PAD];
    __shared__ _Float16 Bs[2][TILE_N][TILE_K + LDS_PAD];

    const int tid  = threadIdx.x;
    const int wave = tid >> 5;
    const int lane = tid & 31;
    const int half = lane >> 4;
    const int r    = lane & 15;

    const int n0 = blockIdx.x * TILE_N;
    const int m0 = blockIdx.y * TILE_M;

    const int wm = (wave & 1) * 64;
    const int wn = (wave >> 1) * 32;

    v8f acc[4][2] = {};

    // ---- prologue: fill buffer 0 with K slice 0 ----
    {
        typename Stage<AT>::ty pa[Stage<AT>::n];
        v4f pb[4];
        tile_ld(X, K, m0, 0, tid, pa);
        tile_ld(W, K, n0, 0, tid, pb);
        tile_st(As[0], tid, pa);
        tile_st(Bs[0], tid, pb);
    }
    __syncthreads();

    const int NK = K / TILE_K;
    for (int kt = 0; kt < NK; ++kt) {
        const int cur   = kt & 1;
        const bool more = (kt + 1) < NK;

        // issue global loads for the next K slice (hidden under the WMMAs)
        typename Stage<AT>::ty sa[Stage<AT>::n];
        v4f sb[4];
        if (more) {
            const int k0n = (kt + 1) * TILE_K;
            tile_ld(X, K, m0, k0n, tid, sa);
            tile_ld(W, K, n0, k0n, tid, sb);
        }

        // A fragments (ISA 16-bit A 16x32 layout): lane (half,r) holds
        // K = half*8 + 0..7 in elems 0..7 and K = 16 + half*8 + 0..7 in 8..15.
        v16h a[4];
#pragma unroll
        for (int i = 0; i < 4; ++i) {
            const int row = wm + i * 16 + r;
            v8h lo = *(const v8h*)&As[cur][row][half * 8];
            v8h hi = *(const v8h*)&As[cur][row][16 + half * 8];
            a[i] = __builtin_shufflevector(lo, hi,
                0,1,2,3,4,5,6,7,8,9,10,11,12,13,14,15);
        }
        // B fragments: lane (half, n=r) holds K = half*8 + 0..7 in elems 0..7 and K = 16 + half*8 + 0..7 in 8..15.
        v16h b[2];
#pragma unroll
        for (int j = 0; j < 2; ++j) {
            const int row = wn + j * 16 + r;
            v8h lo = *(const v8h*)&Bs[cur][row][half * 8];
            v8h hi = *(const v8h*)&Bs[cur][row][16 + half * 8];
            b[j] = __builtin_shufflevector(lo, hi,
                0,1,2,3,4,5,6,7,8,9,10,11,12,13,14,15);
        }

#pragma unroll
        for (int i = 0; i < 4; ++i)
#pragma unroll
            for (int j = 0; j < 2; ++j)
                acc[i][j] = __builtin_amdgcn_wmma_f32_16x16x32_f16(
                    false, a[i], false, b[j], (short)0, acc[i][j], false, false);

        // convert + store next slice into the other buffer
        if (more) {
            const int nxt = cur ^ 1;
            tile_st(As[nxt], tid, sa);
            tile_st(Bs[nxt], tid, sb);
        }
        __syncthreads();
    }

    // Epilogue. D layout: elem e -> M = e + 8*half, N = r.
    const float bv0 = bias[n0 + wn + r];
    const float bv1 = bias[n0 + wn + 16 + r];
#pragma unroll
    for (int i = 0; i < 4; ++i) {
        const int rowBase = m0 + wm + i * 16 + half * 8;
        const int col0    = n0 + wn + r;
#pragma unroll
        for (int e = 0; e < 8; ++e) {
            float v0 = acc[i][0][e] + bv0;
            float v1 = acc[i][1][e] + bv1;
            if (GATED) {
                v0 = 1.0f / (1.0f + __expf(-v0));
                v1 = 1.0f / (1.0f + __expf(-v1));
            }
            Out[(size_t)(rowBase + e) * N + col0]      = (OT)v0;
            Out[(size_t)(rowBase + e) * N + col0 + 16] = (OT)v1;
        }
    }
}

// ---------------------------------------------------------------------------
// Chunked associative scan:  h_t = (1-g_t) h_{t-1} + g_t x_t,  h_{-1} = 0.
// g and h stored as f16 (halves intermediate bandwidth); x is f32.
// Segment decomposition gives B*D*SEG = 524288 independent streams.
// tid = (seg*8 + b)*1024 + d  (consecutive lanes -> consecutive d, coalesced)
// ---------------------------------------------------------------------------
__global__ __launch_bounds__(256)
void scan_pass1(const float* __restrict__ x, const _Float16* __restrict__ g,
                float* __restrict__ segA, float* __restrict__ segH)
{
    const int tid = blockIdx.x * 256 + threadIdx.x;
    const int d   = tid & (DCH - 1);
    const int s   = tid >> 10;          // seg*8 + b
    const int b   = s & (BATCH - 1);
    const int seg = s >> 3;
    const size_t base = ((size_t)b * L_SEQ + (size_t)seg * SEGLEN) * DCH + d;

    float A = 1.0f, H = 0.0f;
#pragma unroll 8
    for (int l = 0; l < SEGLEN; ++l) {
        const size_t idx = base + (size_t)l * DCH;
        const float gv = (float)g[idx];
        const float xv = x[idx];
        const float av = 1.0f - gv;
        H = fmaf(av, H, gv * xv);
        A *= av;
    }
    segA[tid] = A;
    segH[tid] = H;
}

__global__ __launch_bounds__(256)
void scan_pass2(const float* __restrict__ segA, const float* __restrict__ segH,
                float* __restrict__ carry)
{
    const int ch = blockIdx.x * 256 + threadIdx.x;   // b*1024 + d
    float C = 0.0f;
#pragma unroll 4
    for (int seg = 0; seg < SEG; ++seg) {
        const int idx = seg * (BATCH * DCH) + ch;
        carry[idx] = C;                     // carry entering this segment
        C = fmaf(segA[idx], C, segH[idx]); // h at end of this segment
    }
}

__global__ __launch_bounds__(256)
void scan_pass3(const float* __restrict__ x, _Float16* __restrict__ gh, // in: g, out: h
                const float* __restrict__ carry)
{
    const int tid = blockIdx.x * 256 + threadIdx.x;
    const int d   = tid & (DCH - 1);
    const int s   = tid >> 10;
    const int b   = s & (BATCH - 1);
    const int seg = s >> 3;
    const size_t base = ((size_t)b * L_SEQ + (size_t)seg * SEGLEN) * DCH + d;

    float h = carry[tid];
#pragma unroll 8
    for (int l = 0; l < SEGLEN; ++l) {
        const size_t idx = base + (size_t)l * DCH;
        const float gv = (float)gh[idx];
        const float xv = x[idx];
        h = fmaf(1.0f - gv, h, gv * xv);
        gh[idx] = (_Float16)h;              // overwrite g with h in place
    }
}

// ---------------------------------------------------------------------------
extern "C" void kernel_launch(void* const* d_in, const int* in_sizes, int n_in,
                              void* d_out, int out_size, void* d_ws, size_t ws_size,
                              hipStream_t stream)
{
    (void)in_sizes; (void)n_in; (void)out_size; (void)ws_size;
    const float* x  = (const float*)d_in[0];
    const float* Wg = (const float*)d_in[1];
    const float* bg = (const float*)d_in[2];
    const float* Wo = (const float*)d_in[3];
    const float* bo = (const float*)d_in[4];
    float* out = (float*)d_out;

    const int B = BATCH, L = L_SEQ, D = DCH;
    const int M = B * L;                     // 32768

    // workspace layout: f16 g/h buffer, then fp32 segment state
    _Float16* gbuf = (_Float16*)d_ws;                    // [M*D] halves (64MB)
    float* segA  = (float*)(gbuf + (size_t)M * D);       // [B*D*SEG]
    float* segH  = segA + (size_t)B * D * SEG;           // [B*D*SEG]
    float* carry = segH + (size_t)B * D * SEG;           // [B*D*SEG]

    dim3 ggrid(D / TILE_N, M / TILE_M);                  // 8 x 256 (N fast -> L2 reuse)

    // 1) g = sigmoid(x @ Wg^T + bg), stored f16
    gemm_f16_wmma<float, _Float16, true>
        <<<ggrid, 256, 0, stream>>>(x, Wg, bg, gbuf, M, D, D);
    // 2) chunked scan, h overwrites g (f16)
    scan_pass1<<<(B * D * SEG) / 256, 256, 0, stream>>>(x, gbuf, segA, segH);
    scan_pass2<<<(B * D) / 256, 256, 0, stream>>>(segA, segH, carry);
    scan_pass3<<<(B * D * SEG) / 256, 256, 0, stream>>>(x, gbuf, carry);
    // 3) out = h @ Wo^T + bo, h read as f16 directly
    gemm_f16_wmma<_Float16, float, false>
        <<<ggrid, 256, 0, stream>>>(gbuf, Wo, bo, out, M, D, D);
}
